// DeformableConvBlock_23055384445331
// MI455X (gfx1250) — matrix operand order repaired and checked by reading, not compiled, not run on hardware
//
#include <hip/hip_runtime.h>
#include <hip/hip_bf16.h>

// ---------------- problem constants ----------------
#define BATCH 8
#define CIN   128
#define COUT  256
#define HH    64
#define WW    64
#define HW    4096              // HH*WW
#define NPIX  32768             // BATCH*HW
#define KB_MAIN 36              // 1152 / 32
#define KB_SC   4               // 128 / 32
#define EPSV  1e-5f

typedef __attribute__((ext_vector_type(16))) __bf16 v16bf;
typedef __attribute__((ext_vector_type(8)))  float  v8f;

// round-to-nearest-even float -> bf16 bits (no reliance on __bf16 arithmetic)
__device__ __forceinline__ unsigned short f2bf(float f) {
    union { float f; unsigned int u; } cv; cv.f = f;
    unsigned int u = cv.u;
    unsigned int r = u + 0x7FFFu + ((u >> 16) & 1u);
    return (unsigned short)(r >> 16);
}

__device__ __forceinline__ v8f wmma_bf16(v16bf a, v16bf b, v8f c) {
    // D(f32 16x16) = A(bf16 16x32) * B(bf16 32x16) + C
    return __builtin_amdgcn_wmma_f32_16x16x32_bf16(
        /*neg_a=*/false, a, /*neg_b=*/false, b,
        /*c_mod=*/(short)0, c, /*reuse_a=*/false, /*reuse_b=*/false);
}

// K-local index held by (lane, element j) of a bf16 A-fragment (16x32), this code's contiguous order k = 16*(lane/16) + j
__device__ __forceinline__ int a_kloc(int lane, int j) {
    return ((lane < 16) ? 0 : 16) + j;
}

// ================= kernel 0: repack weights into A-fragment layout (bf16) ====
// wA  : [16 mtiles][36 kblocks][32 lanes][16 bf16]   K-order = tap*128 + c
// wAsc: [16 mtiles][ 4 kblocks][32 lanes][16 bf16]   K = c
__global__ __launch_bounds__(256) void k_repack(
    const float* __restrict__ w_def, const float* __restrict__ w_sc,
    unsigned short* __restrict__ wA, unsigned short* __restrict__ wAsc)
{
    const int ent = blockIdx.x * 256 + threadIdx.x;
    if (ent < 16 * KB_MAIN * 32) {
        const int mt   = ent / (KB_MAIN * 32);
        const int r    = ent % (KB_MAIN * 32);
        const int kb   = r >> 5;
        const int lane = r & 31;
        const int m    = mt * 16 + (lane & 15);
#pragma unroll
        for (int j = 0; j < 16; ++j) {
            const int K   = kb * 32 + a_kloc(lane, j);
            const int tap = K >> 7;          // K = tap*128 + c
            const int c   = K & 127;
            wA[(size_t)ent * 16 + j] = f2bf(w_def[((size_t)m * CIN + c) * 9 + tap]);
        }
    } else {
        const int e2 = ent - 16 * KB_MAIN * 32;
        if (e2 < 16 * KB_SC * 32) {
            const int mt   = e2 / (KB_SC * 32);
            const int r    = e2 % (KB_SC * 32);
            const int kb   = r >> 5;
            const int lane = r & 31;
            const int m    = mt * 16 + (lane & 15);
#pragma unroll
            for (int j = 0; j < 16; ++j) {
                const int c = kb * 32 + a_kloc(lane, j);
                wAsc[(size_t)e2 * 16 + j] = f2bf(w_sc[(size_t)m * CIN + c]);
            }
        }
    }
}

// ================= kernel 1: 3x3 offset conv (128 -> 18), f32 ================
__global__ __launch_bounds__(256) void k_offconv(
    const float* __restrict__ x, const float* __restrict__ w_off,
    const float* __restrict__ b_off, float* __restrict__ offs)
{
    const int pix = blockIdx.x * 256 + threadIdx.x;     // 0..NPIX-1
    const int b   = pix >> 12;
    const int rem = pix & 4095;
    const int iy  = rem >> 6;
    const int ix  = rem & 63;

    float acc[18];
#pragma unroll
    for (int o = 0; o < 18; ++o) acc[o] = b_off[o];

    for (int c = 0; c < CIN; ++c) {
        const float* xb = x + ((size_t)(b * CIN + c)) * HW;
        float xv[9];
#pragma unroll
        for (int t = 0; t < 9; ++t) {
            const int yy = iy + t / 3 - 1;
            const int xx = ix + t % 3 - 1;
            xv[t] = (yy >= 0 && yy < HH && xx >= 0 && xx < WW) ? xb[yy * WW + xx] : 0.f;
        }
        const float* wr = w_off + (size_t)c * 9;        // + o*1152 + t
#pragma unroll
        for (int o = 0; o < 18; ++o)
#pragma unroll
            for (int t = 0; t < 9; ++t)
                acc[o] += xv[t] * wr[(size_t)o * 1152 + t];
    }
#pragma unroll
    for (int o = 0; o < 18; ++o)
        offs[((size_t)(b * 18 + o)) * HW + rem] = acc[o];
}

// ====== kernel 2: fused bilinear-sampling im2col + GEMM (WMMA bf16) ==========
// one block = 16-pixel row strip, 8 waves, each wave -> M=32 (all 256 channels)
__global__ __launch_bounds__(256) void k_deform_gemm(
    const float* __restrict__ x, const float* __restrict__ offs,
    const unsigned short* __restrict__ wA, const float* __restrict__ b_def,
    float* __restrict__ out_pre)
{
    __shared__ int4   s_id[9][16];                       // y0c,x0c,y1c,x1c
    __shared__ float4 s_wt[9][16];                       // w00,w01,w10,w11
    __shared__ __align__(32) unsigned short s_B[2][32][16]; // B frag, lane-major

    const int tid  = threadIdx.x;
    const int lane = tid & 31;
    const int wave = tid >> 5;

    const int flat0 = blockIdx.x * 16;
    const int bb    = flat0 >> 12;
    const int rem   = flat0 & 4095;
    const int py    = rem >> 6;
    const int px0   = rem & 63;
    const int prow  = py * WW;

    // ---- per (tap, pixel) bilinear metadata ----
    if (tid < 144) {
        const int t = tid >> 4, n = tid & 15;
        const float dy = offs[((size_t)(bb * 18 + 2 * t    )) * HW + prow + px0 + n];
        const float dx = offs[((size_t)(bb * 18 + 2 * t + 1)) * HW + prow + px0 + n];
        const float pyf = (float)(py + (t / 3) - 1) + dy;
        const float pxf = (float)(px0 + n + (t % 3) - 1) + dx;
        const float y0f = floorf(pyf), x0f = floorf(pxf);
        const float fy = pyf - y0f, fx = pxf - x0f;
        const int y0 = (int)y0f, x0i = (int)x0f;
        const int y1 = y0 + 1,  x1  = x0i + 1;
        const float vy0 = (y0  >= 0 && y0  < HH) ? 1.f : 0.f;
        const float vy1 = (y1  >= 0 && y1  < HH) ? 1.f : 0.f;
        const float vx0 = (x0i >= 0 && x0i < WW) ? 1.f : 0.f;
        const float vx1 = (x1  >= 0 && x1  < WW) ? 1.f : 0.f;
        int4 id; float4 wt;
        id.x = min(max(y0, 0), HH - 1);  id.y = min(max(x0i, 0), WW - 1);
        id.z = min(max(y1, 0), HH - 1);  id.w = min(max(x1,  0), WW - 1);
        wt.x = (1.f - fy) * (1.f - fx) * vy0 * vx0;
        wt.y = (1.f - fy) * fx         * vy0 * vx1;
        wt.z = fy         * (1.f - fx) * vy1 * vx0;
        wt.w = fy         * fx         * vy1 * vx1;
        s_id[t][n] = id; s_wt[t][n] = wt;
    }
    __syncthreads();

    // stage one 32(K) x 16(N) bf16 B-fragment into LDS, already in lane layout
    auto stage = [&](int kb, int buf) {
        const int t  = kb >> 2;                 // tap
        const int c0 = (kb & 3) << 5;           // channel block base
        const int kc = tid >> 3;                // 0..31 local K
        const int nb = (tid & 7) << 1;          // 2 pixels per thread
        const float* xb = x + ((size_t)(bb * CIN + c0 + kc)) * HW;
#pragma unroll
        for (int e = 0; e < 2; ++e) {
            const int n = nb + e;
            const int4   id = s_id[t][n];
            const float4 wt = s_wt[t][n];
            const float v = wt.x * xb[id.x * WW + id.y]
                          + wt.y * xb[id.x * WW + id.w]
                          + wt.z * xb[id.z * WW + id.y]
                          + wt.w * xb[id.z * WW + id.w];
            const int ln = (kc < 16) ? n : (16 + n);
            s_B[buf][ln][kc & 15] = f2bf(v);
        }
    };

    v8f acc0 = {}; v8f acc1 = {};
    const int mt0 = wave * 2, mt1 = wave * 2 + 1;

    stage(0, 0);
    __syncthreads();
    for (int kb = 0; kb < KB_MAIN; ++kb) {
        if (kb + 1 < KB_MAIN) {
            __builtin_prefetch(wA + ((size_t)(mt0 * KB_MAIN + kb + 1) * 32 + lane) * 16, 0, 1);
            stage(kb + 1, (kb + 1) & 1);
        }
        const v16bf bfrag = *(const v16bf*)(&s_B[kb & 1][lane][0]);
        const v16bf a0 = *(const v16bf*)(wA + ((size_t)(mt0 * KB_MAIN + kb) * 32 + lane) * 16);
        const v16bf a1 = *(const v16bf*)(wA + ((size_t)(mt1 * KB_MAIN + kb) * 32 + lane) * 16);
        acc0 = wmma_bf16(a0, bfrag, acc0);
        acc1 = wmma_bf16(a1, bfrag, acc1);
        __syncthreads();
    }

    // epilogue: + b_def, store pre-BN activations
    const int col = lane & 15;
    const int hi  = (lane >= 16) ? 8 : 0;
#pragma unroll
    for (int i = 0; i < 8; ++i) {
        int ch = mt0 * 16 + i + hi;
        out_pre[((size_t)(bb * COUT + ch)) * HW + prow + px0 + col] = acc0[i] + b_def[ch];
        ch = mt1 * 16 + i + hi;
        out_pre[((size_t)(bb * COUT + ch)) * HW + prow + px0 + col] = acc1[i] + b_def[ch];
    }
}

// ========== kernel 3: per-channel mean / inv-std (deterministic) =============
__global__ __launch_bounds__(256) void k_stats(
    const float* __restrict__ out_pre, float* __restrict__ meanv, float* __restrict__ istdv)
{
    __shared__ float s_s[256], s_q[256];
    const int ch = blockIdx.x, tid = threadIdx.x;
    float s = 0.f, q = 0.f;
    for (int b = 0; b < BATCH; ++b) {
        const float* p = out_pre + ((size_t)(b * COUT + ch)) * HW;
        for (int i = tid; i < HW; i += 256) { const float v = p[i]; s += v; q += v * v; }
    }
    s_s[tid] = s; s_q[tid] = q; __syncthreads();
    for (int o = 128; o > 0; o >>= 1) {
        if (tid < o) { s_s[tid] += s_s[tid + o]; s_q[tid] += s_q[tid + o]; }
        __syncthreads();
    }
    if (tid == 0) {
        const float m   = s_s[0] * (1.f / (float)NPIX);
        const float var = s_q[0] * (1.f / (float)NPIX) - m * m;
        meanv[ch] = m;
        istdv[ch] = rsqrtf(var + EPSV);
    }
}

// ==== kernel 4: BN + 1x1 shortcut GEMM (WMMA bf16) + residual + ReLU =========
__global__ __launch_bounds__(256) void k_final(
    const float* __restrict__ x, const float* __restrict__ out_pre,
    const unsigned short* __restrict__ wAsc,
    const float* __restrict__ meanv, const float* __restrict__ istdv,
    const float* __restrict__ gamma, const float* __restrict__ beta,
    const float* __restrict__ b_sc, float* __restrict__ out)
{
    __shared__ __align__(32) unsigned short s_X[KB_SC][32][16];

    const int tid  = threadIdx.x;
    const int lane = tid & 31;
    const int wave = tid >> 5;

    const int flat0 = blockIdx.x * 16;
    const int bb    = flat0 >> 12;
    const int rem   = flat0 & 4095;
    const int prow  = (rem >> 6) * WW;
    const int px0   = rem & 63;

    // stage x tile (128 c x 16 n) into B-fragment layout
    {
        const int c  = tid >> 1;            // 0..127
        const int n0 = (tid & 1) << 3;      // 8 pixels each
        const float* xr = x + ((size_t)(bb * CIN + c)) * HW + prow + px0;
        const int kb = c >> 5, kc = c & 31;
#pragma unroll
        for (int e = 0; e < 8; ++e) {
            const int n  = n0 + e;
            const int ln = (kc < 16) ? n : (16 + n);
            s_X[kb][ln][kc & 15] = f2bf(xr[n]);
        }
    }
    __syncthreads();

    v8f acc0 = {}, acc1 = {};
    const int mt0 = wave * 2, mt1 = mt0 + 1;
#pragma unroll
    for (int kb = 0; kb < KB_SC; ++kb) {
        const v16bf bfrag = *(const v16bf*)(&s_X[kb][lane][0]);
        const v16bf a0 = *(const v16bf*)(wAsc + ((size_t)(mt0 * KB_SC + kb) * 32 + lane) * 16);
        const v16bf a1 = *(const v16bf*)(wAsc + ((size_t)(mt1 * KB_SC + kb) * 32 + lane) * 16);
        acc0 = wmma_bf16(a0, bfrag, acc0);
        acc1 = wmma_bf16(a1, bfrag, acc1);
    }

    const int col = lane & 15;
    const int hi  = (lane >= 16) ? 8 : 0;
#pragma unroll
    for (int i = 0; i < 8; ++i) {
        {
            const int ch = mt0 * 16 + i + hi;
            const size_t idx = ((size_t)(bb * COUT + ch)) * HW + prow + px0 + col;
            const float bn = (out_pre[idx] - meanv[ch]) * istdv[ch] * gamma[ch] + beta[ch];
            out[idx] = fmaxf(bn + acc0[i] + b_sc[ch], 0.f);
        }
        {
            const int ch = mt1 * 16 + i + hi;
            const size_t idx = ((size_t)(bb * COUT + ch)) * HW + prow + px0 + col;
            const float bn = (out_pre[idx] - meanv[ch]) * istdv[ch] * gamma[ch] + beta[ch];
            out[idx] = fmaxf(bn + acc1[i] + b_sc[ch], 0.f);
        }
    }
}

// =============================== host launcher ===============================
extern "C" void kernel_launch(void* const* d_in, const int* in_sizes, int n_in,
                              void* d_out, int out_size, void* d_ws, size_t ws_size,
                              hipStream_t stream)
{
    (void)in_sizes; (void)n_in; (void)out_size; (void)ws_size;
    const float* x     = (const float*)d_in[0];
    const float* w_off = (const float*)d_in[1];
    const float* b_off = (const float*)d_in[2];
    const float* w_def = (const float*)d_in[3];
    const float* b_def = (const float*)d_in[4];
    const float* gamma = (const float*)d_in[5];
    const float* beta  = (const float*)d_in[6];
    const float* w_sc  = (const float*)d_in[7];
    const float* b_sc  = (const float*)d_in[8];
    float* out = (float*)d_out;

    auto align_up = [](size_t v) { return (v + 255) & ~(size_t)255; };
    char* p = (char*)d_ws;
    float* offs    = (float*)p; p += align_up((size_t)BATCH * 18   * HW * sizeof(float));
    float* out_pre = (float*)p; p += align_up((size_t)BATCH * COUT * HW * sizeof(float));
    float* meanv   = (float*)p; p += align_up(COUT * sizeof(float));
    float* istdv   = (float*)p; p += align_up(COUT * sizeof(float));
    unsigned short* wA   = (unsigned short*)p; p += align_up((size_t)16 * KB_MAIN * 32 * 16 * 2);
    unsigned short* wAsc = (unsigned short*)p; p += align_up((size_t)16 * KB_SC   * 32 * 16 * 2);

    // repack: 16*36*32 + 16*4*32 = 20480 entries
    k_repack<<<80, 256, 0, stream>>>(w_def, w_sc, wA, wAsc);
    // offset conv: one thread per pixel
    k_offconv<<<NPIX / 256, 256, 0, stream>>>(x, w_off, b_off, offs);
    // fused deformable GEMM: one block per 16-pixel strip
    k_deform_gemm<<<NPIX / 16, 256, 0, stream>>>(x, offs, wA, b_def, out_pre);
    // batch stats: one block per output channel
    k_stats<<<COUT, 256, 0, stream>>>(out_pre, meanv, istdv);
    // BN + shortcut GEMM + ReLU
    k_final<<<NPIX / 16, 256, 0, stream>>>(x, out_pre, wAsc, meanv, istdv,
                                           gamma, beta, b_sc, out);
}
